// Mamba_Encoder_Layer_42649025249790
// MI455X (gfx1250) — matrix operand order repaired and checked by reading, not compiled, not run on hardware
//
#include <hip/hip_runtime.h>
#include <hip/hip_bf16.h>

// ---------------- problem constants ----------------
#define D_MODEL_C 512
#define D_FF_C    2048
#define D_STATE_C 16
#define D_CONV_C  4
#define D_INNER_C 1024
#define DT_RANK_C 33
#define B_C       2
#define L_C       2048
#define M_C       (B_C * L_C)     // 4096 rows

typedef __bf16 bf16_t;
typedef __attribute__((ext_vector_type(16))) __bf16 bf16x16;
typedef __attribute__((ext_vector_type(8)))  __bf16 bf16x8;
typedef __attribute__((ext_vector_type(8)))  float  f32x8;
typedef unsigned int u32x4 __attribute__((ext_vector_type(4)));
typedef int          i32x4 __attribute__((ext_vector_type(4)));
typedef int          i32x8 __attribute__((ext_vector_type(8)));

// ---------------- f32 -> bf16 conversion ----------------
__global__ void cvt_f32_bf16(const float* __restrict__ src, bf16_t* __restrict__ dst, int n) {
    int i = blockIdx.x * blockDim.x + threadIdx.x;
    if (i < n) dst[i] = (bf16_t)src[i];
}

// ---------------- Tensor Data Mover: 2D bf16 tile (rows x 32) global -> LDS ----------------
// D# per CDNA5 ISA: group0 = {count/lds_addr/global_addr/type=2},
// group1 = {data_size=2B, tensor_dim0=32, tensor_dim1=rows, tile 32 x rows, stride0=ld}.
__device__ __forceinline__ void tdm_load_tile_2d(unsigned int lds_off, const bf16_t* src,
                                                 int rows, int ld_elems)
{
    unsigned long long ga = (unsigned long long)(uintptr_t)src;
    u32x4 g0;
    g0[0] = 1u;                                                    // count=1, user mode
    g0[1] = lds_off;                                               // LDS byte address
    g0[2] = (unsigned int)(ga & 0xffffffffu);                      // global_addr[31:0]
    g0[3] = (unsigned int)((ga >> 32) & 0x01ffffffu) | (2u << 30); // addr[56:32] | type=2
    unsigned int td0 = 32u;
    unsigned int td1 = (unsigned int)rows;
    unsigned int st0 = (unsigned int)ld_elems;
    i32x8 g1;
    g1[0] = (int)(1u << 16);                                       // data_size=1 (2 bytes)
    g1[1] = (int)((td0 & 0xffffu) << 16);                          // tensor_dim0 lo
    g1[2] = (int)(((td0 >> 16) & 0xffffu) | ((td1 & 0xffffu) << 16));
    g1[3] = (int)(((td1 >> 16) & 0xffffu) | (32u << 16));          // tile_dim0 = 32
    g1[4] = (int)(td1 & 0xffffu);                                  // tile_dim1 = rows, tile_dim2 = 0
    g1[5] = (int)st0;                                              // tensor_dim0_stride lo
    g1[6] = 0;
    g1[7] = 0;
    i32x4 gz = {0, 0, 0, 0};
#if defined(__clang_major__) && (__clang_major__ >= 23)
    i32x8 gz8 = {0, 0, 0, 0, 0, 0, 0, 0};
    __builtin_amdgcn_tensor_load_to_lds(g0, g1, gz, gz, gz8, 0);
#else
    __builtin_amdgcn_tensor_load_to_lds(g0, g1, gz, gz, 0);
#endif
}

// ---------------- TDM-staged bf16 WMMA GEMM ----------------
// C[M,N] = act(A[M,K] * B[N,K]^T + bias); A,B bf16 row-major [*,K].
// Block 256 threads = 8 waves; block tile 128x128; wave tile 32x64 (2x4 WMMA accums).
// Double-buffered LDS tiles filled by the Tensor Data Mover (wave 0 issues).
template <int HAS_BIAS, int ACT_RELU, int OUT_BF16>
__global__ __launch_bounds__(256)
void gemm_wmma_tdm(const bf16_t* __restrict__ A, const bf16_t* __restrict__ Bw,
                   const float* __restrict__ bias, void* __restrict__ Cout,
                   int M, int N, int K)
{
    constexpr int BM = 128, BN = 128, BK = 32;
    __shared__ __align__(16) bf16_t lA[2][BM * BK];
    __shared__ __align__(16) bf16_t lB[2][BN * BK];

    const int lane = threadIdx.x & 31;
    const int wave = threadIdx.x >> 5;   // 0..7
    const int wm   = wave >> 1;          // 0..3 -> 32-row strip
    const int wn   = wave & 1;           // 0..1 -> 64-col strip
    const int block_m = blockIdx.x * BM;
    const int block_n = blockIdx.y * BN;
    const int lm = lane & 15;
    const int kh = lane >> 4;

    f32x8 acc[2][4] = {};

    const int nk = K / BK;
    if (wave == 0) {
        tdm_load_tile_2d((unsigned int)(uintptr_t)&lA[0][0], A + (size_t)block_m * K, BM, K);
        tdm_load_tile_2d((unsigned int)(uintptr_t)&lB[0][0], Bw + (size_t)block_n * K, BN, K);
    }

    for (int kt = 0; kt < nk; ++kt) {
        const int buf = kt & 1;
        if (wave == 0) {
            if (kt + 1 < nk) {
                const int k0n = (kt + 1) * BK;
                tdm_load_tile_2d((unsigned int)(uintptr_t)&lA[buf ^ 1][0],
                                 A + (size_t)block_m * K + k0n, BM, K);
                tdm_load_tile_2d((unsigned int)(uintptr_t)&lB[buf ^ 1][0],
                                 Bw + (size_t)block_n * K + k0n, BN, K);
                __builtin_amdgcn_s_wait_tensorcnt((short)2);   // current pair done (in-order)
            } else {
                __builtin_amdgcn_s_wait_tensorcnt((short)0);
            }
        }
        __syncthreads();   // current buffer visible to all waves

        bf16x16 afrag[2], bfrag[4];
        #pragma unroll
        for (int i = 0; i < 2; ++i) {
            const int row = wm * 32 + i * 16 + lm;
            bf16x8 lo = *(const bf16x8*)&lA[buf][row * BK + kh * 8];
            bf16x8 hi = *(const bf16x8*)&lA[buf][row * BK + 16 + kh * 8];
            #pragma unroll
            for (int e = 0; e < 8; ++e) { afrag[i][e] = lo[e]; afrag[i][e + 8] = hi[e]; }
        }
        #pragma unroll
        for (int j = 0; j < 4; ++j) {
            const int col = wn * 64 + j * 16 + lm;
            bf16x8 lo = *(const bf16x8*)&lB[buf][col * BK + kh * 8];
            bf16x8 hi = *(const bf16x8*)&lB[buf][col * BK + 16 + kh * 8];
            #pragma unroll
            for (int e = 0; e < 8; ++e) { bfrag[j][e] = lo[e]; bfrag[j][e + 8] = hi[e]; }
        }
        #pragma unroll
        for (int i = 0; i < 2; ++i)
            #pragma unroll
            for (int j = 0; j < 4; ++j)
                acc[i][j] = __builtin_amdgcn_wmma_f32_16x16x32_bf16(
                    false, afrag[i], false, bfrag[j], (short)0, acc[i][j], false, false);

        __syncthreads();   // all reads done before buffer is refilled
    }

    // Epilogue: VGPR r -> m = base + (lane/16)*8 + r ; n = lane%16
    #pragma unroll
    for (int i = 0; i < 2; ++i) {
        const int mbase = block_m + wm * 32 + i * 16 + kh * 8;
        #pragma unroll
        for (int j = 0; j < 4; ++j) {
            const int n = block_n + wn * 64 + j * 16 + lm;
            float bv = 0.0f;
            if (HAS_BIAS) bv = bias[n];
            #pragma unroll
            for (int r = 0; r < 8; ++r) {
                float v = acc[i][j][r];
                if (HAS_BIAS) v += bv;
                if (ACT_RELU) v = fmaxf(v, 0.0f);
                size_t idx = (size_t)(mbase + r) * N + n;
                if (OUT_BF16) ((bf16_t*)Cout)[idx] = (bf16_t)v;
                else          ((float*)Cout)[idx]  = v;
            }
        }
    }
}

// ---------------- depthwise causal conv (k=4) + SiLU ----------------
__global__ void conv_silu_kernel(const float* __restrict__ xz, const float* __restrict__ Wc,
                                 const float* __restrict__ bc, float* __restrict__ xc)
{
    int t = blockIdx.x * blockDim.x + threadIdx.x;
    if (t >= M_C * D_INNER_C) return;
    int d = t & (D_INNER_C - 1);
    int m = t >> 10;
    int l = m & (L_C - 1);
    int b = m >> 11;
    float acc = bc[d];
    const float* w = Wc + d * D_CONV_C;
    #pragma unroll
    for (int j = 0; j < D_CONV_C; ++j) {
        int ls = l - (D_CONV_C - 1) + j;
        if (ls >= 0)
            acc += w[j] * xz[(size_t)(b * L_C + ls) * (2 * D_INNER_C) + d];
    }
    xc[t] = acc / (1.0f + __expf(-acc));  // SiLU
}

// ---------------- x_proj: x_dbl[m, 0..64] = xc[m,:] . W_xproj[n,:] ----------------
__global__ void xdbl_kernel(const float* __restrict__ xc, const float* __restrict__ Wx,
                            float* __restrict__ xdbl)
{
    int t = blockIdx.x * blockDim.x + threadIdx.x;
    const int NX = DT_RANK_C + 2 * D_STATE_C;   // 65
    if (t >= M_C * NX) return;
    int n = t % NX;
    int m = t / NX;
    const float* a = xc + (size_t)m * D_INNER_C;
    const float* w = Wx + (size_t)n * D_INNER_C;
    float acc = 0.0f;
    for (int k = 0; k < D_INNER_C; ++k) acc += a[k] * w[k];
    xdbl[t] = acc;
}

// ---------------- dt = softplus(dtr . W_dt^T + b_dt) ----------------
__global__ void dt_kernel(const float* __restrict__ xdbl, const float* __restrict__ Wdt,
                          const float* __restrict__ bdt, float* __restrict__ dt)
{
    int t = blockIdx.x * blockDim.x + threadIdx.x;
    if (t >= M_C * D_INNER_C) return;
    int d = t & (D_INNER_C - 1);
    int m = t >> 10;
    const float* a = xdbl + (size_t)m * (DT_RANK_C + 2 * D_STATE_C); // dtr = first 33
    const float* w = Wdt + (size_t)d * DT_RANK_C;
    float acc = bdt[d];
    #pragma unroll 11
    for (int k = 0; k < DT_RANK_C; ++k) acc += a[k] * w[k];
    dt[t] = (acc > 20.0f) ? acc : __logf(1.0f + __expf(acc));  // softplus
}

// ---------------- selective scan + skip + gating, store y as bf16 ----------------
// lane group of 16 = one (b,d) channel; lane within group = state s.
__global__ void scan_kernel(const float* __restrict__ dt, const float* __restrict__ xc,
                            const float* __restrict__ xdbl, const float* __restrict__ A_log,
                            const float* __restrict__ Dp, const float* __restrict__ xz,
                            bf16_t* __restrict__ ybf)
{
    int t = blockIdx.x * blockDim.x + threadIdx.x;   // 0 .. B*D_INNER*16
    if (t >= B_C * D_INNER_C * D_STATE_C) return;
    const int s  = t & (D_STATE_C - 1);
    const int ch = t >> 4;
    const int d  = ch & (D_INNER_C - 1);
    const int b  = ch >> 10;
    const int NX = DT_RANK_C + 2 * D_STATE_C;        // 65

    const float a  = -__expf(A_log[d * D_STATE_C + s]);
    const float Dv = Dp[d];
    float h = 0.0f;

    for (int l = 0; l < L_C; ++l) {
        const int m = b * L_C + l;
        const float dtv = dt[(size_t)m * D_INNER_C + d];
        const float xcv = xc[(size_t)m * D_INNER_C + d];
        const float Bv  = xdbl[(size_t)m * NX + DT_RANK_C + s];
        const float Cv  = xdbl[(size_t)m * NX + DT_RANK_C + D_STATE_C + s];
        h = __expf(dtv * a) * h + dtv * xcv * Bv;
        float p = h * Cv;
        #pragma unroll
        for (int off = 8; off > 0; off >>= 1) p += __shfl_xor(p, off, 32);
        if (s == 0) {
            float y = p + xcv * Dv;
            float z = xz[(size_t)m * (2 * D_INNER_C) + D_INNER_C + d];
            y *= z / (1.0f + __expf(-z));            // y * silu(z)
            ybf[(size_t)m * D_INNER_C + d] = (bf16_t)y;
        }
    }
}

// ---------------- launch ----------------
extern "C" void kernel_launch(void* const* d_in, const int* in_sizes, int n_in,
                              void* d_out, int out_size, void* d_ws, size_t ws_size,
                              hipStream_t stream)
{
    const float* x       = (const float*)d_in[0];
    const float* W_in    = (const float*)d_in[1];
    const float* W_conv  = (const float*)d_in[2];
    const float* b_conv  = (const float*)d_in[3];
    const float* W_xproj = (const float*)d_in[4];
    const float* W_dt    = (const float*)d_in[5];
    const float* b_dt    = (const float*)d_in[6];
    const float* A_log   = (const float*)d_in[7];
    const float* Dp      = (const float*)d_in[8];
    const float* W_out   = (const float*)d_in[9];
    const float* W1      = (const float*)d_in[10];
    const float* b1      = (const float*)d_in[11];
    const float* W2      = (const float*)d_in[12];
    const float* b2      = (const float*)d_in[13];
    float* out = (float*)d_out;

    char* p = (char*)d_ws;
    auto alloc = [&](size_t bytes) -> void* {
        void* r = (void*)p;
        p += (bytes + 255) & ~(size_t)255;
        return r;
    };
    bf16_t* xbf    = (bf16_t*)alloc((size_t)M_C * D_MODEL_C * 2);
    bf16_t* Winbf  = (bf16_t*)alloc((size_t)2 * D_INNER_C * D_MODEL_C * 2);
    bf16_t* Woutbf = (bf16_t*)alloc((size_t)D_MODEL_C * D_INNER_C * 2);
    bf16_t* W1bf   = (bf16_t*)alloc((size_t)D_FF_C * D_MODEL_C * 2);
    bf16_t* W2bf   = (bf16_t*)alloc((size_t)D_MODEL_C * D_FF_C * 2);
    float*  xz     = (float*)alloc((size_t)M_C * 2 * D_INNER_C * 4);
    float*  xc     = (float*)alloc((size_t)M_C * D_INNER_C * 4);
    float*  xdbl   = (float*)alloc((size_t)M_C * (DT_RANK_C + 2 * D_STATE_C) * 4);
    float*  dt     = (float*)alloc((size_t)M_C * D_INNER_C * 4);
    bf16_t* ybf    = (bf16_t*)alloc((size_t)M_C * D_INNER_C * 2);
    bf16_t* y2bf   = (bf16_t*)alloc((size_t)M_C * D_MODEL_C * 2);
    bf16_t* h1bf   = (bf16_t*)alloc((size_t)M_C * D_FF_C * 2);

    const int TB = 256;
    auto cvt = [&](const float* s, bf16_t* dd, int n) {
        cvt_f32_bf16<<<(n + TB - 1) / TB, TB, 0, stream>>>(s, dd, n);
    };
    cvt(x,     xbf,    M_C * D_MODEL_C);
    cvt(W_in,  Winbf,  2 * D_INNER_C * D_MODEL_C);
    cvt(W_out, Woutbf, D_MODEL_C * D_INNER_C);
    cvt(W1,    W1bf,   D_FF_C * D_MODEL_C);
    cvt(W2,    W2bf,   D_MODEL_C * D_FF_C);

    // 1) xz = x @ W_in^T   [4096 x 2048], K=512
    gemm_wmma_tdm<0, 0, 0><<<dim3(M_C / 128, (2 * D_INNER_C) / 128), 256, 0, stream>>>(
        xbf, Winbf, nullptr, xz, M_C, 2 * D_INNER_C, D_MODEL_C);

    // 2) depthwise conv + SiLU -> xc
    conv_silu_kernel<<<(M_C * D_INNER_C + TB - 1) / TB, TB, 0, stream>>>(xz, W_conv, b_conv, xc);

    // 3) x_dbl = xc @ W_xproj^T  [4096 x 65]
    {
        int n = M_C * (DT_RANK_C + 2 * D_STATE_C);
        xdbl_kernel<<<(n + TB - 1) / TB, TB, 0, stream>>>(xc, W_xproj, xdbl);
    }

    // 4) dt = softplus(dtr @ W_dt^T + b_dt)
    dt_kernel<<<(M_C * D_INNER_C + TB - 1) / TB, TB, 0, stream>>>(xdbl, W_dt, b_dt, dt);

    // 5) selective scan + skip + gating -> ybf (bf16)
    scan_kernel<<<(B_C * D_INNER_C * D_STATE_C) / TB, TB, 0, stream>>>(
        dt, xc, xdbl, A_log, Dp, xz, ybf);

    // 6) y2 = y @ W_out^T  [4096 x 512], K=1024 -> bf16
    gemm_wmma_tdm<0, 0, 1><<<dim3(M_C / 128, D_MODEL_C / 128), 256, 0, stream>>>(
        ybf, Woutbf, nullptr, y2bf, M_C, D_MODEL_C, D_INNER_C);

    // 7) h1 = relu(y2 @ W1^T + b1)  [4096 x 2048], K=512 -> bf16
    gemm_wmma_tdm<1, 1, 1><<<dim3(M_C / 128, D_FF_C / 128), 256, 0, stream>>>(
        y2bf, W1bf, b1, h1bf, M_C, D_FF_C, D_MODEL_C);

    // 8) out = h1 @ W2^T + b2  [4096 x 512], K=2048 -> f32
    gemm_wmma_tdm<1, 0, 0><<<dim3(M_C / 128, D_MODEL_C / 128), 256, 0, stream>>>(
        h1bf, W2bf, b2, out, M_C, D_MODEL_C, D_FF_C);

    (void)in_sizes; (void)n_in; (void)out_size; (void)ws_size;
}
